// ScaledDotProductAttention_48335561949315
// MI455X (gfx1250) — matrix operand order repaired and checked by reading, not compiled, not run on hardware
//
#include <hip/hip_runtime.h>

// ScaledDotProductAttention for MI455X (gfx1250, wave32, WMMA bf16).
// B=2 H=16 S=2048 D=64. Outputs: context [B,H,S,64] f32 then weights [B,H,S,S] f32.

#define S_LEN 2048
#define D_HEAD 64
#define NKT (S_LEN / 16)      // 128 k-tiles
#define WAVES 8
#define SC_STRIDE 2052        // padded LDS row stride (floats) -> shifts banks by 4/row
#define CTX_STRIDE 68         // padded context row stride (floats)

typedef __attribute__((ext_vector_type(16))) __bf16 v16bf;
typedef __attribute__((ext_vector_type(8)))  float  v8f;

union BF16x16 { v16bf v; unsigned u[8]; };

// pack two f32 into one dword of two bf16 (truncate) -- single v_perm_b32
__device__ __forceinline__ unsigned pkbf(float lo, float hi) {
  return __builtin_amdgcn_perm(__float_as_uint(hi), __float_as_uint(lo), 0x07060302u);
}

__global__ __launch_bounds__(256)
void sdpa_wmma_bf16_kernel(const float* __restrict__ Q, const float* __restrict__ K,
                           const float* __restrict__ V, const int* __restrict__ mask,
                           float* __restrict__ ctx_out, float* __restrict__ w_out)
{
  extern __shared__ float smem[];
  float* s_sc  = smem;                    // [16][SC_STRIDE] scores -> probs
  float* s_ctx = smem + 16 * SC_STRIDE;   // [16][CTX_STRIDE] context accumulator

  const int tid  = threadIdx.x;
  const int lane = tid & 31;
  const int w    = tid >> 5;              // wave id 0..7
  const int qt   = blockIdx.x;            // q-tile
  const int bh   = blockIdx.y;            // fused batch*head
  const int q0   = qt * 16;

  const int rsel = lane & 15;             // A row / B,C column selector
  const int hsel = lane >> 4;             // lane-half selector
  const int rowh = hsel * 8;              // C row base for this half

  const size_t head_off = (size_t)bh * S_LEN * D_HEAD;
  const float* Qh = Q + head_off;
  const float* Kh = K + head_off;
  const float* Vh = V + head_off;

  // ---- Q A-tiles (16x32 bf16, K-halves 0..31 / 32..63), kept in VGPRs ----
  // A layout: lane m=rsel; lanes<16 hold K 0..15, lanes>=16 hold K 16..31
  v16bf qa[2];
  {
    const float* qrow = Qh + (size_t)(q0 + rsel) * D_HEAD + hsel * 16;
    #pragma unroll
    for (int h = 0; h < 2; ++h) {
      const float* p = qrow + h * 32;
      BF16x16 t;
      #pragma unroll
      for (int i = 0; i < 4; ++i) t.u[i]     = pkbf(p[2 * i],      p[2 * i + 1]);
      #pragma unroll
      for (int i = 0; i < 4; ++i) t.u[4 + i] = pkbf(p[8 + 2 * i], p[8 + 2 * i + 1]);
      qa[h] = t.v;
    }
  }

  // zero the shared context accumulator
  for (int i = tid; i < 16 * CTX_STRIDE; i += 256) s_ctx[i] = 0.0f;

  // ---- Phase 1: scores = mask(scale(Q K^T)) into LDS ----
  const int* mbase = mask + ((size_t)bh * S_LEN + q0) * S_LEN;
  for (int kt = w; kt < NKT; kt += WAVES) {
    // branchless prefetch of this wave's next K tile
    const int pkt = (kt + WAVES < NKT) ? (kt + WAVES) : kt;
    __builtin_prefetch(Kh + (size_t)(pkt * 16 + rsel) * D_HEAD, 0, 0);

    // B = K^T (64x16): lane col n=rsel -> K row kt*16+n; lanes<16 K-dim 0..15, lanes>=16 16..31
    const float* krow = Kh + (size_t)(kt * 16 + rsel) * D_HEAD + hsel * 16;
    v16bf kb[2];
    #pragma unroll
    for (int h = 0; h < 2; ++h) {
      const float* p = krow + h * 32;
      BF16x16 t;
      #pragma unroll
      for (int i = 0; i < 8; ++i) t.u[i] = pkbf(p[2 * i], p[2 * i + 1]);
      kb[h] = t.v;
    }
    v8f c = {};
    c = __builtin_amdgcn_wmma_f32_16x16x32_bf16(false, qa[0], false, kb[0], (short)0, c, false, false);
    c = __builtin_amdgcn_wmma_f32_16x16x32_bf16(false, qa[1], false, kb[1], (short)0, c, false, false);

    const int col = kt * 16 + rsel;
    const int* mp = mbase + col;
    #pragma unroll
    for (int j = 0; j < 8; ++j) {          // C vgpr j -> row j (+8 for upper lanes)
      const int r  = j + rowh;
      const int mv = mp[(size_t)r * S_LEN];
      const float v = c[j] * 0.125f;       // 1/sqrt(64)
      s_sc[r * SC_STRIDE + col] = mv ? -1e9f : v;
    }
  }
  __syncthreads();

  // ---- Phase 2: per-row softmax; weights -> global, probs -> LDS in place ----
  float* wbase = w_out + ((size_t)bh * S_LEN + q0) * S_LEN;
  #pragma unroll
  for (int rr = 0; rr < 2; ++rr) {
    const int r = w * 2 + rr;
    float* row = s_sc + r * SC_STRIDE;
    float m = -3.4e38f;
    for (int i = lane; i < S_LEN; i += 32) m = fmaxf(m, row[i]);
    #pragma unroll
    for (int off = 16; off > 0; off >>= 1) m = fmaxf(m, __shfl_xor(m, off, 32));
    float sum = 0.0f;
    for (int i = lane; i < S_LEN; i += 32) {
      float e = __expf(row[i] - m);
      row[i] = e;
      sum += e;
    }
    #pragma unroll
    for (int off = 16; off > 0; off >>= 1) sum += __shfl_xor(sum, off, 32);
    const float inv = 1.0f / sum;
    float* wg = wbase + (size_t)r * S_LEN;
    for (int i = lane; i < S_LEN; i += 32) {
      const float p = row[i] * inv;
      row[i] = p;
      wg[i]  = p;                           // fully coalesced 128B stores per wave
    }
  }
  __syncthreads();

  // ---- Phase 3: context += P @ V via bf16 WMMA ----
  v8f acc[4] = {};                          // N = 64 = 4 tiles of 16
  for (int cc = w; cc < S_LEN / 32; cc += WAVES) {   // 32-wide K chunks
    // A tile: P[0..15][cc*32 .. +31] from LDS f32 -> bf16
    const float* pr = s_sc + rsel * SC_STRIDE + cc * 32 + hsel * 16;
    BF16x16 ta;
    #pragma unroll
    for (int i = 0; i < 4; ++i) ta.u[i]     = pkbf(pr[2 * i],      pr[2 * i + 1]);
    #pragma unroll
    for (int i = 0; i < 4; ++i) ta.u[4 + i] = pkbf(pr[8 + 2 * i], pr[8 + 2 * i + 1]);
    // B tiles from V (32x16 each): lane col fixed, walk K rows (coalesced across lanes)
    const float* vp = Vh + (size_t)(cc * 32 + hsel * 16) * D_HEAD + rsel;
    #pragma unroll
    for (int nt = 0; nt < 4; ++nt) {
      const float* p = vp + nt * 16;
      BF16x16 tb;
      #pragma unroll
      for (int i = 0; i < 8; ++i)
        tb.u[i] = pkbf(p[(2 * i) * D_HEAD], p[(2 * i + 1) * D_HEAD]);
      acc[nt] = __builtin_amdgcn_wmma_f32_16x16x32_bf16(false, ta.v, false, tb.v,
                                                        (short)0, acc[nt], false, false);
    }
  }
  // cross-wave reduction of partial context
  #pragma unroll
  for (int nt = 0; nt < 4; ++nt) {
    #pragma unroll
    for (int j = 0; j < 8; ++j)
      atomicAdd(&s_ctx[(j + rowh) * CTX_STRIDE + nt * 16 + rsel], acc[nt][j]);
  }
  __syncthreads();

  // ---- Phase 4: write context tile (16x64), b128 coalesced ----
  {
    const int e = tid * 4;                  // 1024 elements / 256 threads
    const int r = e >> 6, d = e & 63;
    float4 v = *(const float4*)&s_ctx[r * CTX_STRIDE + d];
    *(float4*)&ctx_out[((size_t)(bh * S_LEN + q0 + r)) * D_HEAD + d] = v;
  }
}

extern "C" void kernel_launch(void* const* d_in, const int* in_sizes, int n_in,
                              void* d_out, int out_size, void* d_ws, size_t ws_size,
                              hipStream_t stream) {
  const float* Q    = (const float*)d_in[0];
  const float* K    = (const float*)d_in[1];
  const float* V    = (const float*)d_in[2];
  const int*   mask = (const int*)d_in[3];

  const int BH = in_sizes[0] / (S_LEN * D_HEAD);          // 32
  float* ctx = (float*)d_out;                              // [BH, S, 64]
  float* wts = (float*)d_out + (size_t)BH * S_LEN * D_HEAD; // [BH, S, S]

  const size_t shmem = (size_t)(16 * SC_STRIDE + 16 * CTX_STRIDE) * sizeof(float); // ~136 KB
  hipFuncSetAttribute((const void*)sdpa_wmma_bf16_kernel,
                      hipFuncAttributeMaxDynamicSharedMemorySize, (int)shmem);

  dim3 grid(NKT, BH);   // 128 q-tiles x 32 heads
  sdpa_wmma_bf16_kernel<<<grid, 256, shmem, stream>>>(Q, K, V, mask, ctx, wts);
}
